// PLA_Attention_Model_25829933318888
// MI455X (gfx1250) — matrix operand order repaired and checked by reading, not compiled, not run on hardware
//
#include <hip/hip_runtime.h>
#include <hip/hip_bf16.h>

typedef __attribute__((ext_vector_type(16))) _Float16 v16h;
typedef __attribute__((ext_vector_type(8)))  _Float16 v8h;
typedef __attribute__((ext_vector_type(8)))  float    v8f;

namespace {
constexpr int Pn = 128, Bn = 32, Sn = 64, En = 50, EP = 64;
constexpr int Hn = 128, G3 = 384, KP = 192;     // 3H gates, K = EP(64) + H(128)
constexpr int PB = Pn * Bn;                     // 4096
constexpr int NCLS = 11;
constexpr int MBLK = 128, NBLK = 64;
}

__device__ __forceinline__ float sigmf(float x) { return 1.0f / (1.0f + __expf(-x)); }

// ---- pack [Wih | 0-pad | Whh] into f16, layout (2, G3, KP) ----
__global__ void pack_w_kernel(const float* __restrict__ Wih, const float* __restrict__ Whh,
                              _Float16* __restrict__ Wout) {
  int idx = blockIdx.x * blockDim.x + threadIdx.x;
  int total = 2 * G3 * KP;
  if (idx >= total) return;
  int k = idx % KP;
  int g = (idx / KP) % G3;
  int d = idx / (KP * G3);
  float v = 0.0f;
  if (k < En)        v = Wih[((size_t)d * G3 + g) * En + k];
  else if (k >= EP)  v = Whh[((size_t)d * G3 + g) * Hn + (k - EP)];
  Wout[idx] = (_Float16)v;
}

// ---- embedding gather -> f16 padded (Sn, PB, EP) ----
__global__ void gather_embed_kernel(const int* __restrict__ flow, const float* __restrict__ emb,
                                    _Float16* __restrict__ X) {
  size_t idx = (size_t)blockIdx.x * blockDim.x + threadIdx.x;
  size_t total = (size_t)Sn * PB * EP;
  if (idx >= total) return;
  int k = (int)(idx % EP);
  int n = (int)((idx / EP) % PB);
  int s = (int)(idx / ((size_t)EP * PB));
  int p = n / Bn, b = n % Bn;
  int tok = flow[((size_t)p * Bn + b) * Sn + s];
  float v = (k < En) ? emb[(size_t)tok * En + k] : 0.0f;
  X[idx] = (_Float16)v;
}

// ---- h0 init: byte level, h0_bytes (P,2,B,H) -> hF/h16 (2, PB, H) ----
__global__ void init_h_byte_kernel(const float* __restrict__ h0, float* __restrict__ hF,
                                   _Float16* __restrict__ h16) {
  int idx = blockIdx.x * blockDim.x + threadIdx.x;
  int total = 2 * PB * Hn;
  if (idx >= total) return;
  int j = idx % Hn;
  int n = (idx / Hn) % PB;
  int d = idx / (PB * Hn);
  int p = n / Bn, b = n % Bn;
  float v = h0[(((size_t)p * 2 + d) * Bn + b) * Hn + j];
  hF[idx] = v;
  h16[idx] = (_Float16)v;
}

// ---- h0 init: packet level, h0_packet (2,B,H) direct ----
__global__ void init_h_pkt_kernel(const float* __restrict__ h0, float* __restrict__ hF,
                                  _Float16* __restrict__ h16) {
  int idx = blockIdx.x * blockDim.x + threadIdx.x;
  int total = 2 * Bn * Hn;
  if (idx >= total) return;
  float v = h0[idx];
  hF[idx] = v;
  h16[idx] = (_Float16)v;
}

// ---- fused GRU-step GEMM: G = [x_s | h] @ W^T with separate x/h accumulators ----
// Double-buffered LDS, tiles staged with CDNA5 GLOBAL_LOAD_ASYNC_TO_LDS_B128 (ASYNCcnt).
// X   : (Ssz, M, EP) f16   H16 : (2, M, Hn) f16   W16 : (2, G3, KP) f16
// Gx,Gh : (2, M, G3) f32   grid = (ceil(M/128), G3/64, 2 dirs), block = 256 (8 waves)
__global__ void gru_gemm_kernel(const _Float16* __restrict__ X,
                                const _Float16* __restrict__ H16,
                                const _Float16* __restrict__ W16,
                                float* __restrict__ Gx, float* __restrict__ Gh,
                                int M, int s, int Ssz) {
  const int d  = blockIdx.z;
  const int m0 = blockIdx.x * MBLK;
  const int g0 = blockIdx.y * NBLK;
  const int sx = (d == 0) ? s : (Ssz - 1 - s);
  const _Float16* Xs = X   + (size_t)sx * M * EP;
  const _Float16* Hs = H16 + (size_t)d * M * Hn;
  const _Float16* Ws = W16 + (size_t)d * G3 * KP;

  __shared__ __align__(16) _Float16 ldsA[2][MBLK][32];
  __shared__ __align__(16) _Float16 ldsB[2][NBLK][32];

  const int tid  = threadIdx.x;
  const int lane = tid & 31;
  const int wave = tid >> 5;
  const int wm = wave & 3;    // 32-row band within 128
  const int wn = wave >> 2;   // 32-col band within 64

  // Every wave issues exactly 3 async-load instructions per stage (2 for A, 1 for B);
  // out-of-range A rows are clamped (data discarded by the guarded store), keeping
  // EXEC uniform and the ASYNCcnt arithmetic exact.
  auto prefetch = [&](int kc, int buf) {
#pragma unroll
    for (int it = 0; it < 2; ++it) {
      int seg = tid + it * 256;
      int r = seg >> 2;                 // LDS row 0..127
      int c = (seg & 3) * 8;            // half offset 0,8,16,24 (16B granules)
      int gr = m0 + r;
      int grc = (gr < M) ? gr : 0;      // clamp: keeps loads in-bounds & uniform
      const _Float16* base = (kc < 2) ? Xs : Hs;
      int rowStride = (kc < 2) ? EP : Hn;
      int kofs = (kc < 2) ? (kc * 32) : ((kc - 2) * 32);
      unsigned voff = (unsigned)(((size_t)grc * rowStride + kofs + c) * sizeof(_Float16));
      unsigned ldsa = (unsigned)(size_t)&ldsA[buf][r][c];
      asm volatile("global_load_async_to_lds_b128 %0, %1, %2"
                   :: "v"(ldsa), "v"(voff), "s"(base) : "memory");
    }
    {
      int n = tid >> 2;
      int c = (tid & 3) * 8;
      unsigned voff = (unsigned)(((size_t)(g0 + n) * KP + kc * 32 + c) * sizeof(_Float16));
      unsigned ldsb = (unsigned)(size_t)&ldsB[buf][n][c];
      asm volatile("global_load_async_to_lds_b128 %0, %1, %2"
                   :: "v"(ldsb), "v"(voff), "s"(Ws) : "memory");
    }
  };

  v8f accx[2][2], acch[2][2];
  {
    v8f z = {};
#pragma unroll
    for (int mi = 0; mi < 2; ++mi)
#pragma unroll
      for (int ni = 0; ni < 2; ++ni) { accx[mi][ni] = z; acch[mi][ni] = z; }
  }

  prefetch(0, 0);

#pragma unroll
  for (int kc = 0; kc < 6; ++kc) {           // K chunks of 32: 0-1 = x part, 2-5 = h part
    const int cur = kc & 1;
    if (kc < 5) {
      prefetch(kc + 1, cur ^ 1);
      // own wave issued 3 more; <=3 outstanding means current stage has landed
      asm volatile("s_wait_asynccnt 3" ::: "memory");
    } else {
      asm volatile("s_wait_asynccnt 0" ::: "memory");
    }
    __syncthreads();                          // all waves' async writes visible

    const int laneLo = lane & 15;
    const int laneHi = lane >> 4;
    v16h afrag[2], bfrag[2];
    // A fragment (16x32): lanes 0-15 K=0..7,16..23; lanes 16-31 K=8..15,24..31
#pragma unroll
    for (int mi = 0; mi < 2; ++mi) {
      int row = wm * 32 + mi * 16 + laneLo;
      v8h lo = *(const v8h*)&ldsA[cur][row][laneHi * 8];
      v8h hi = *(const v8h*)&ldsA[cur][row][16 + laneHi * 8];
      afrag[mi] = __builtin_shufflevector(lo, hi, 0,1,2,3,4,5,6,7,8,9,10,11,12,13,14,15);
    }
    // B fragment (32x16): lanes 0-15 K=0..7,16..23; lanes 16-31 K=8..15,24..31 (row of W^T = gate col)
#pragma unroll
    for (int ni = 0; ni < 2; ++ni) {
      int row = wn * 32 + ni * 16 + laneLo;
      v8h lo = *(const v8h*)&ldsB[cur][row][laneHi * 8];
      v8h hi = *(const v8h*)&ldsB[cur][row][16 + laneHi * 8];
      bfrag[ni] = __builtin_shufflevector(lo, hi, 0,1,2,3,4,5,6,7,8,9,10,11,12,13,14,15);
    }
#pragma unroll
    for (int mi = 0; mi < 2; ++mi)
#pragma unroll
      for (int ni = 0; ni < 2; ++ni) {
        if (kc < 2)
          accx[mi][ni] = __builtin_amdgcn_wmma_f32_16x16x32_f16(
              false, afrag[mi], false, bfrag[ni], (short)0, accx[mi][ni], false, false);
        else
          acch[mi][ni] = __builtin_amdgcn_wmma_f32_16x16x32_f16(
              false, afrag[mi], false, bfrag[ni], (short)0, acch[mi][ni], false, false);
      }
    __syncthreads();                          // safe to overwrite buf[cur] next stage
  }

  // store C: VGPR r -> M = r + 8*(lane>>4), N = lane&15
  const size_t base = (size_t)d * M * G3;
  const int laneN = lane & 15;
  const int laneMhi = (lane >> 4) * 8;
#pragma unroll
  for (int mi = 0; mi < 2; ++mi)
#pragma unroll
    for (int ni = 0; ni < 2; ++ni)
#pragma unroll
      for (int r = 0; r < 8; ++r) {
        int row = m0 + wm * 32 + mi * 16 + laneMhi + r;
        int col = g0 + wn * 32 + ni * 16 + laneN;
        if (row < M) {
          size_t o = base + (size_t)row * G3 + col;
          Gx[o] = accx[mi][ni][r];
          Gh[o] = acch[mi][ni][r];
        }
      }
}

// ---- GRU gate update: r,z use Gx+Gh; n uses Gx_n + r*Gh_n (PyTorch order r,z,n) ----
__global__ void gru_gate_kernel(const float* __restrict__ Gx, const float* __restrict__ Gh,
                                float* __restrict__ hF, _Float16* __restrict__ h16,
                                _Float16* __restrict__ Y, int M, int s, int Ssz) {
  int idx = blockIdx.x * blockDim.x + threadIdx.x;
  int total = 2 * M * Hn;
  if (idx >= total) return;
  int j = idx % Hn;
  int n = (idx / Hn) % M;
  int d = idx / (M * Hn);
  size_t gb = ((size_t)d * M + n) * G3;
  float r  = sigmf(Gx[gb + j] + Gh[gb + j]);
  float z  = sigmf(Gx[gb + Hn + j] + Gh[gb + Hn + j]);
  float nn = tanhf(Gx[gb + 2 * Hn + j] + r * Gh[gb + 2 * Hn + j]);
  size_t hb = ((size_t)d * M + n) * Hn + j;
  float h = (1.0f - z) * nn + z * hF[hb];
  hF[hb]  = h;
  h16[hb] = (_Float16)h;
  int sy = (d == 0) ? s : (Ssz - 1 - s);   // backward outputs un-reversed
  Y[((size_t)sy * M + n) * (2 * Hn) + d * Hn + j] = (_Float16)h;
}

// ---- byte-level attention + encoder projection; one 64-thread block per (p,b) ----
__global__ void byte_attn_kernel(const _Float16* __restrict__ Y,  // (Sn, PB, 2H)
                                 const float* __restrict__ hF,    // (2, PB, H) final
                                 const float* __restrict__ attnW, // (E, 2H)
                                 const float* __restrict__ attnB, // (E)
                                 _Float16* __restrict__ Xp) {     // (Pn*Bn, EP)
  const int pb = blockIdx.x;
  const int p = pb / Bn, b = pb % Bn;
  const int t = threadIdx.x;
  __shared__ float hv[2 * Hn];
  __shared__ float sc[Sn];
  __shared__ float ctx[2 * Hn];
  __shared__ float red[64];

  // hn_p via raw C-order reshape of per-packet (2,B,H): rows 2b, 2b+1 of [hf;hb]
  for (int j = t; j < 2 * Hn; j += 64) {
    int rowi = 2 * b + (j >= Hn ? 1 : 0);
    int jj = j & (Hn - 1);
    float v = (rowi < Bn)
        ? hF[((size_t)(p * Bn + rowi)) * Hn + jj]
        : hF[((size_t)PB + p * Bn + (rowi - Bn)) * Hn + jj];
    hv[j] = v;
  }
  __syncthreads();
  {
    const _Float16* yr = Y + ((size_t)t * PB + pb) * (2 * Hn);
    float a = 0.0f;
    for (int j = 0; j < 2 * Hn; ++j) a += (float)yr[j] * hv[j];
    sc[t] = a;
  }
  __syncthreads();
  red[t] = sc[t]; __syncthreads();
  for (int o = 32; o > 0; o >>= 1) { if (t < o) red[t] = fmaxf(red[t], red[t + o]); __syncthreads(); }
  float mx = red[0]; __syncthreads();
  float e = __expf(sc[t] - mx);
  red[t] = e; __syncthreads();
  for (int o = 32; o > 0; o >>= 1) { if (t < o) red[t] += red[t + o]; __syncthreads(); }
  float inv = 1.0f / red[0]; __syncthreads();
  sc[t] = e * inv;
  __syncthreads();
  for (int j = t; j < 2 * Hn; j += 64) {
    float a = 0.0f;
    for (int s = 0; s < Sn; ++s) a += sc[s] * (float)Y[((size_t)s * PB + pb) * (2 * Hn) + j];
    ctx[j] = a;
  }
  __syncthreads();
  for (int ei = t; ei < EP; ei += 64) {
    _Float16 outv = (_Float16)0.0f;
    if (ei < En) {
      float a = attnB[ei];
      for (int j = 0; j < 2 * Hn; ++j) a += attnW[(size_t)ei * (2 * Hn) + j] * ctx[j];
      outv = (_Float16)a;
    }
    Xp[(size_t)pb * EP + ei] = outv;
  }
}

// ---- packet-level attention + classifier; one 64-thread block per batch b ----
__global__ void pkt_attn_cls_kernel(const _Float16* __restrict__ Yp, // (Pn, Bn, 2H)
                                    const float* __restrict__ hFp,   // (2, Bn, H)
                                    const float* __restrict__ attnW, // (Po, 2H)
                                    const float* __restrict__ attnB,
                                    const float* __restrict__ clsW,  // (NCLS, Po)
                                    const float* __restrict__ clsB,
                                    float* __restrict__ out) {
  const int b = blockIdx.x;
  const int t = threadIdx.x;
  __shared__ float hv[2 * Hn];
  __shared__ float sc[Pn];
  __shared__ float ctx[2 * Hn];
  __shared__ float ov[Hn];
  __shared__ float red[64];

  for (int j = t; j < 2 * Hn; j += 64) {          // hn2.reshape(B, 2H) semantics
    int rowi = 2 * b + (j >= Hn ? 1 : 0);
    hv[j] = hFp[(size_t)rowi * Hn + (j & (Hn - 1))];
  }
  __syncthreads();
  for (int pi = t; pi < Pn; pi += 64) {
    const _Float16* yr = Yp + ((size_t)pi * Bn + b) * (2 * Hn);
    float a = 0.0f;
    for (int j = 0; j < 2 * Hn; ++j) a += (float)yr[j] * hv[j];
    sc[pi] = a;
  }
  __syncthreads();
  float m = -3.4e38f;
  for (int pi = t; pi < Pn; pi += 64) m = fmaxf(m, sc[pi]);
  red[t] = m; __syncthreads();
  for (int o = 32; o > 0; o >>= 1) { if (t < o) red[t] = fmaxf(red[t], red[t + o]); __syncthreads(); }
  float mx = red[0]; __syncthreads();
  float ssum = 0.0f;
  for (int pi = t; pi < Pn; pi += 64) { float e = __expf(sc[pi] - mx); sc[pi] = e; ssum += e; }
  red[t] = ssum; __syncthreads();
  for (int o = 32; o > 0; o >>= 1) { if (t < o) red[t] += red[t + o]; __syncthreads(); }
  float inv = 1.0f / red[0]; __syncthreads();
  for (int j = t; j < 2 * Hn; j += 64) {
    float a = 0.0f;
    for (int pi = 0; pi < Pn; ++pi) a += sc[pi] * (float)Yp[((size_t)pi * Bn + b) * (2 * Hn) + j];
    ctx[j] = a * inv;
  }
  __syncthreads();
  for (int k = t; k < Hn; k += 64) {
    float a = attnB[k];
    for (int j = 0; j < 2 * Hn; ++j) a += attnW[(size_t)k * (2 * Hn) + j] * ctx[j];
    ov[k] = a;
  }
  __syncthreads();
  if (t < NCLS) {
    float a = clsB[t];
    for (int k = 0; k < Hn; ++k) a += clsW[(size_t)t * Hn + k] * ov[k];
    out[(size_t)b * NCLS + t] = a;
  }
}

extern "C" void kernel_launch(void* const* d_in, const int* in_sizes, int n_in,
                              void* d_out, int out_size, void* d_ws, size_t ws_size,
                              hipStream_t stream) {
  (void)in_sizes; (void)n_in; (void)out_size; (void)ws_size;
  const int*   flow      = (const int*)  d_in[0];
  const float* emb       = (const float*)d_in[1];
  const float* byte_Wih  = (const float*)d_in[2];
  const float* byte_Whh  = (const float*)d_in[3];
  const float* byte_aW   = (const float*)d_in[4];
  const float* byte_aB   = (const float*)d_in[5];
  const float* pkt_Wih   = (const float*)d_in[6];
  const float* pkt_Whh   = (const float*)d_in[7];
  const float* pkt_aW    = (const float*)d_in[8];
  const float* pkt_aB    = (const float*)d_in[9];
  const float* cls_W     = (const float*)d_in[10];
  const float* cls_b     = (const float*)d_in[11];
  const float* h0_bytes  = (const float*)d_in[12];
  const float* h0_packet = (const float*)d_in[13];
  float* out = (float*)d_out;

  char* ws = (char*)d_ws;
  size_t off = 0;
  auto alloc = [&](size_t bytes) -> void* {
    void* p = ws + off;
    off += (bytes + 255) & ~(size_t)255;
    return p;
  };
  _Float16* Wb16 = (_Float16*)alloc((size_t)2 * G3 * KP * 2);
  _Float16* Wp16 = (_Float16*)alloc((size_t)2 * G3 * KP * 2);
  _Float16* Xb   = (_Float16*)alloc((size_t)Sn * PB * EP * 2);
  float*    hF   = (float*)   alloc((size_t)2 * PB * Hn * 4);
  _Float16* h16  = (_Float16*)alloc((size_t)2 * PB * Hn * 2);
  float*    Gx   = (float*)   alloc((size_t)2 * PB * G3 * 4);
  float*    Gh   = (float*)   alloc((size_t)2 * PB * G3 * 4);
  _Float16* Yb   = (_Float16*)alloc((size_t)Sn * PB * 2 * Hn * 2);
  _Float16* Xp   = (_Float16*)alloc((size_t)Pn * Bn * EP * 2);
  float*    hFp  = (float*)   alloc((size_t)2 * Bn * Hn * 4);
  _Float16* h16p = (_Float16*)alloc((size_t)2 * Bn * Hn * 2);
  float*    Gpx  = (float*)   alloc((size_t)2 * Bn * G3 * 4);
  float*    Gph  = (float*)   alloc((size_t)2 * Bn * G3 * 4);
  _Float16* Yp   = (_Float16*)alloc((size_t)Pn * Bn * 2 * Hn * 2);

  {
    int total = 2 * G3 * KP;
    int nb = (total + 255) / 256;
    pack_w_kernel<<<nb, 256, 0, stream>>>(byte_Wih, byte_Whh, Wb16);
    pack_w_kernel<<<nb, 256, 0, stream>>>(pkt_Wih,  pkt_Whh,  Wp16);
  }
  {
    size_t total = (size_t)Sn * PB * EP;
    gather_embed_kernel<<<(int)((total + 255) / 256), 256, 0, stream>>>(flow, emb, Xb);
  }
  {
    int total = 2 * PB * Hn;
    init_h_byte_kernel<<<(total + 255) / 256, 256, 0, stream>>>(h0_bytes, hF, h16);
  }
  // byte-level bi-GRU: 64 sequential steps, both directions per launch
  dim3 gB(PB / MBLK, G3 / NBLK, 2);
  for (int s = 0; s < Sn; ++s) {
    gru_gemm_kernel<<<gB, 256, 0, stream>>>(Xb, h16, Wb16, Gx, Gh, PB, s, Sn);
    int total = 2 * PB * Hn;
    gru_gate_kernel<<<(total + 255) / 256, 256, 0, stream>>>(Gx, Gh, hF, h16, Yb, PB, s, Sn);
  }
  byte_attn_kernel<<<PB, 64, 0, stream>>>(Yb, hF, byte_aW, byte_aB, Xp);
  {
    int total = 2 * Bn * Hn;
    init_h_pkt_kernel<<<(total + 255) / 256, 256, 0, stream>>>(h0_packet, hFp, h16p);
  }
  // packet-level bi-GRU: 128 sequential steps over packets, M = 32
  dim3 gP((Bn + MBLK - 1) / MBLK, G3 / NBLK, 2);
  for (int p = 0; p < Pn; ++p) {
    gru_gemm_kernel<<<gP, 256, 0, stream>>>(Xp, h16p, Wp16, Gpx, Gph, Bn, p, Pn);
    int total = 2 * Bn * Hn;
    gru_gate_kernel<<<(total + 255) / 256, 256, 0, stream>>>(Gpx, Gph, hFp, h16p, Yp, Bn, p, Pn);
  }
  pkt_attn_cls_kernel<<<Bn, 64, 0, stream>>>(Yp, hFp, pkt_aW, pkt_aB, cls_W, cls_b, out);
}
